// VoxelGNNGenerator_66546223284347
// MI455X (gfx1250) — compile-verified, matrix operand order checked by reading, not run on hardware
//
#include <hip/hip_runtime.h>
#include <hip/hip_bf16.h>

typedef __attribute__((ext_vector_type(16))) _Float16 v16h;
typedef __attribute__((ext_vector_type(8)))  _Float16 v8h;
typedef __attribute__((ext_vector_type(8)))  float    v8f;

#define NEG_SLOPE 0.2f
#define EPS 1e-5f
#define NUM_TYPES 8
#define ASTR 40   // LDS row stride in halves (16B-aligned, padded)

union HF { v16h v; v8h h[2]; };

// ---------------------------------------------------------------------------
// Tiled WMMA GEMM:  C[M,Cout] (f32) = A[M,Kpad] (f16) x Wt[Cout,Kpad]^T + bias
// Block tile 64x64, 4 waves, each wave owns a 32x32 quadrant (2x2 wmma accs).
// Double-buffered LDS, register prefetch, single barrier per K-step.
// ---------------------------------------------------------------------------
__global__ __launch_bounds__(128)
void gemm_wmma_f16(const _Float16* __restrict__ A, int lda,
                   const _Float16* __restrict__ Bt, int ldb,
                   const float* __restrict__ bias,
                   float* __restrict__ C, int ldc,
                   int M, int Kpad)
{
    __shared__ __align__(16) _Float16 As[2][64 * ASTR];
    __shared__ __align__(16) _Float16 Bs[2][64 * ASTR];

    const int tid  = threadIdx.x;
    const int wave = tid >> 5;
    const int lane = tid & 31;
    const int rowBase = blockIdx.y * 64;
    const int colBase = blockIdx.x * 64;

    v8f acc[2][2] = {};

    // ---- cooperative loader mapping: thread -> (row, 16-half chunk) ----
    const int lrow = tid >> 1;         // 0..63
    const int loff = (tid & 1) * 16;   // 0 or 16 (halves)
    int grA = rowBase + lrow;
    if (grA >= M) grA = M - 1;         // branchless tail clamp (store side guards)
    const _Float16* aptr = A  + (size_t)grA * lda + loff;
    const _Float16* bptr = Bt + (size_t)(colBase + lrow) * ldb + loff;

    // ---- fragment addressing (ISA 16-bit A/B layouts) ----
    const int rq  = (wave >> 1) * 32;
    const int cq  = (wave & 1) * 32;
    const int l15 = lane & 15;
    const int g   = lane >> 4;

    // ---- prefetch tile 0 into registers ----
    v8h pa0 = *(const v8h*)(aptr);
    v8h pa1 = *(const v8h*)(aptr + 8);
    v8h pb0 = *(const v8h*)(bptr);
    v8h pb1 = *(const v8h*)(bptr + 8);

    int buf = 0;
    for (int k0 = 0; k0 < Kpad; k0 += 32) {
        // ---- commit prefetched tile to LDS[buf] ----
        _Float16* as = As[buf];
        _Float16* bs = Bs[buf];
        *(v8h*)&as[lrow * ASTR + loff]     = pa0;
        *(v8h*)&as[lrow * ASTR + loff + 8] = pa1;
        *(v8h*)&bs[lrow * ASTR + loff]     = pb0;
        *(v8h*)&bs[lrow * ASTR + loff + 8] = pb1;
        __syncthreads();

        // ---- issue next-tile global loads (latency hidden behind WMMA) ----
        int kn = k0 + 32;
        if (kn > Kpad - 32) kn = Kpad - 32;   // final-iter clamp; result unused
        pa0 = *(const v8h*)(aptr + kn);
        pa1 = *(const v8h*)(aptr + kn + 8);
        pb0 = *(const v8h*)(bptr + kn);
        pb1 = *(const v8h*)(bptr + kn + 8);

        // ---- fragment loads + 4 x wmma ----
        HF af[2], bf[2];
        #pragma unroll
        for (int i = 0; i < 2; ++i) {
            int r = rq + i * 16 + l15;
            af[i].h[0] = *(const v8h*)&as[r * ASTR + g * 8];
            af[i].h[1] = *(const v8h*)&as[r * ASTR + 16 + g * 8];
            int c = cq + i * 16 + l15;
            bf[i].h[0] = *(const v8h*)&bs[c * ASTR + g * 8];
            bf[i].h[1] = *(const v8h*)&bs[c * ASTR + 16 + g * 8];
        }
        #pragma unroll
        for (int i = 0; i < 2; ++i)
            #pragma unroll
            for (int j = 0; j < 2; ++j)
                acc[i][j] = __builtin_amdgcn_wmma_f32_16x16x32_f16(
                    false, af[i].v, false, bf[j].v,
                    (short)0, acc[i][j], false, false);
        buf ^= 1;
        // next iteration's ds_store targets the other buffer; the barrier above
        // (next iteration) orders reuse of this buffer after everyone's compute.
    }

    // ---- store D (lanes 0-15: rows g*8+0..7, lanes 16-31: rows 8..15) ----
    #pragma unroll
    for (int i = 0; i < 2; ++i) {
        #pragma unroll
        for (int j = 0; j < 2; ++j) {
            int n = colBase + cq + j * 16 + l15;
            float bz = bias ? bias[n] : 0.0f;
            int mbase = rowBase + rq + i * 16 + g * 8;
            #pragma unroll
            for (int r = 0; r < 8; ++r) {
                int m = mbase + r;
                if (m < M) C[(size_t)m * ldc + n] = acc[i][j][r] + bz;
            }
        }
    }
}

// ---------------------------------------------------------------------------
// Per-column sum / sum-of-squares (block owns 64 columns -> no atomics)
// ---------------------------------------------------------------------------
__global__ __launch_bounds__(256)
void col_reduce(const float* __restrict__ Y, int C, int M,
                float* __restrict__ sum, float* __restrict__ sq)
{
    int c = blockIdx.x * 64 + (threadIdx.x & 63);
    int rl = threadIdx.x >> 6;   // 0..3
    float s = 0.f, q = 0.f;
    if (c < C)
        for (int m = rl; m < M; m += 4) {
            float v = Y[(size_t)m * C + c];
            s += v; q += v * v;
        }
    __shared__ float ls[256], lq[256];
    ls[threadIdx.x] = s; lq[threadIdx.x] = q;
    __syncthreads();
    if (rl == 0 && c < C) {
        int t = threadIdx.x;
        sum[c] = ls[t] + ls[t + 64] + ls[t + 128] + ls[t + 192];
        sq[c]  = lq[t] + lq[t + 64] + lq[t + 128] + lq[t + 192];
    }
}

// ---------------------------------------------------------------------------
// Fused BatchNorm+LeakyReLU (mode 0) or GraphNorm+ReLU (mode 1), f16 out(s)
// ---------------------------------------------------------------------------
__global__ __launch_bounds__(256)
void norm_act(const float* __restrict__ Y, int C, int M,
              const float* __restrict__ sum, const float* __restrict__ sq,
              const float* __restrict__ g, const float* __restrict__ be,
              const float* __restrict__ alpha, int mode,
              _Float16* __restrict__ out1, int ld1, int off1,
              _Float16* __restrict__ out2, int ld2, int off2)
{
    int i = blockIdx.x * 256 + threadIdx.x;
    if (i >= M * C) return;
    int m = i / C, c = i - m * C;
    float inv = 1.0f / (float)M;
    float mu = sum[c] * inv;
    float e2 = sq[c] * inv;
    float y = Y[i];
    float o;
    if (mode == 0) {
        float var = e2 - mu * mu;
        float xn = (y - mu) * rsqrtf(var + EPS);
        o = xn * g[c] + be[c];
        o = o > 0.f ? o : NEG_SLOPE * o;
    } else {
        float a = alpha[c];
        float var = e2 + (a * a - 2.f * a) * mu * mu;  // E[(y-a*mu)^2]
        o = g[c] * (y - a * mu) * rsqrtf(var + EPS) + be[c];
        o = o > 0.f ? o : 0.f;
    }
    out1[(size_t)m * ld1 + off1 + c] = (_Float16)o;
    if (out2) out2[(size_t)m * ld2 + off2 + c] = (_Float16)o;
}

// ---------------------------------------------------------------------------
// Weight convert + transpose: Wt[n*Kpad+k] = (k<K) ? W[k*N+n] : 0
// ---------------------------------------------------------------------------
__global__ __launch_bounds__(256)
void wt_convert(const float* __restrict__ W, _Float16* __restrict__ Wt,
                int K, int Nout, int Kpad)
{
    int i = blockIdx.x * 256 + threadIdx.x;
    if (i >= Nout * Kpad) return;
    int n = i / Kpad, k = i - n * Kpad;
    Wt[i] = (k < K) ? (_Float16)W[(size_t)k * Nout + n] : (_Float16)0.f;
}

// ---------------------------------------------------------------------------
// GCN pieces
// ---------------------------------------------------------------------------
__global__ void deg_init(int* deg, int n) {
    int i = blockIdx.x * 256 + threadIdx.x;
    if (i < n) deg[i] = 1;  // self-loop
}
__global__ void deg_accum(const int* __restrict__ dst, int* __restrict__ deg, int ne) {
    int i = blockIdx.x * 256 + threadIdx.x;
    if (i < ne) atomicAdd(&deg[dst[i]], 1);
}
__global__ void deg_finish(const int* __restrict__ deg, float* __restrict__ dis, int n) {
    int i = blockIdx.x * 256 + threadIdx.x;
    if (i < n) dis[i] = rsqrtf((float)deg[i]);
}
__global__ __launch_bounds__(256)
void gcn_self(const float* __restrict__ H, const float* __restrict__ dis,
              const float* __restrict__ bias, float* __restrict__ Out, int C, int M)
{
    int i = blockIdx.x * 256 + threadIdx.x;
    if (i >= M * C) return;
    int m = i / C, c = i - m * C;
    float d = dis[m];
    Out[i] = d * d * H[i] + bias[c];
}
__global__ __launch_bounds__(256)
void gcn_scatter(const float* __restrict__ H, const int* __restrict__ src,
                 const int* __restrict__ dst, const float* __restrict__ dis,
                 float* __restrict__ Out, int C)
{
    int e = blockIdx.x;
    int c = blockIdx.y * 256 + threadIdx.x;
    if (c >= C) return;
    int s = src[e], d = dst[e];
    float nw = dis[s] * dis[d];
    const float* hp = H + (size_t)s * C;
    __builtin_prefetch(hp + c, 0, 0);   // global_prefetch_b8
    atomicAdd(&Out[(size_t)d * C + c], nw * hp[c]);
}

// ---------------------------------------------------------------------------
// Type means over local graph (120 nodes, 13 dims, 8 types) — one block
// ---------------------------------------------------------------------------
__global__ void type_mean(const float* __restrict__ lx, const int* __restrict__ lt,
                          float* __restrict__ tm, int nl)
{
    __shared__ float s[NUM_TYPES][13];
    __shared__ int cnt[NUM_TYPES];
    int t = threadIdx.x;
    if (t < NUM_TYPES * 13) ((float*)s)[t] = 0.f;
    if (t < NUM_TYPES) cnt[t] = 0;
    __syncthreads();
    for (int i = t; i < nl; i += 128) {
        int ty = lt[i];
        atomicAdd(&cnt[ty], 1);
        for (int k = 0; k < 13; ++k) atomicAdd(&s[ty][k], lx[i * 13 + k]);
    }
    __syncthreads();
    if (t < NUM_TYPES * 13) {
        int ty = t / 13, k = t - ty * 13;
        tm[t] = cnt[ty] > 0 ? s[ty][k] / (float)cnt[ty] : 0.f;
    }
}
__global__ void cnt_init(int* c) { if (threadIdx.x < NUM_TYPES) c[threadIdx.x] = 0; }
__global__ void vox_count(const int* __restrict__ vt, int* __restrict__ c, int n) {
    int i = blockIdx.x * 256 + threadIdx.x;
    if (i < n) atomicAdd(&c[vt[i]], 1);
}

// ---------------------------------------------------------------------------
// Local encoder on the 8 distinct type rows; BN stats are count-weighted.
// One block, 128 threads (thread = output column).
// ---------------------------------------------------------------------------
__device__ __forceinline__ void lge_bn_act(float* col, const float* w8,
                                           const float* g, const float* be, int c,
                                           float (*h)[128])
{
    float mu = 0.f, e2 = 0.f;
    for (int t = 0; t < NUM_TYPES; ++t) { mu += w8[t] * col[t]; e2 += w8[t] * col[t] * col[t]; }
    float is = rsqrtf(e2 - mu * mu + EPS);
    for (int t = 0; t < NUM_TYPES; ++t) {
        float o = (col[t] - mu) * is * g[c] + be[c];
        o = o > 0.f ? o : NEG_SLOPE * o;
        col[t] = o; h[t][c] = o;
    }
}

__global__ __launch_bounds__(128)
void lge_forward(const float* __restrict__ tm, const int* __restrict__ cnt, int Ntot,
    const float* W0, const float* b0, const float* g0, const float* be0,
    const float* W1, const float* b1, const float* g1, const float* be1,
    const float* W2, const float* b2, const float* g2, const float* be2,
    _Float16* __restrict__ enc)
{
    __shared__ float h[NUM_TYPES][128];
    __shared__ float w8[NUM_TYPES];
    int c = threadIdx.x;
    if (c < NUM_TYPES) w8[c] = (float)cnt[c] / (float)Ntot;
    __syncthreads();

    float col[NUM_TYPES];
    for (int t = 0; t < NUM_TYPES; ++t) {
        float a = b0[c];
        for (int k = 0; k < 13; ++k) a += tm[t * 13 + k] * W0[k * 128 + c];
        col[t] = a;
    }
    lge_bn_act(col, w8, g0, be0, c, h);
    __syncthreads();

    for (int t = 0; t < NUM_TYPES; ++t) {
        float a = b1[c];
        for (int k = 0; k < 128; ++k) a += h[t][k] * W1[k * 128 + c];
        col[t] = a;
    }
    __syncthreads();
    lge_bn_act(col, w8, g1, be1, c, h);
    __syncthreads();

    for (int t = 0; t < NUM_TYPES; ++t) {
        float a = b2[c];
        for (int k = 0; k < 128; ++k) a += h[t][k] * W2[k * 128 + c];
        col[t] = a;
    }
    __syncthreads();
    lge_bn_act(col, w8, g2, be2, c, h);
    for (int t = 0; t < NUM_TYPES; ++t) enc[t * 128 + c] = (_Float16)col[t];
}

// ---------------------------------------------------------------------------
// Concat builders (write f16, zero-pad tail columns)
// ---------------------------------------------------------------------------
__global__ __launch_bounds__(256)
void build_cat0(const _Float16* __restrict__ enc, const int* __restrict__ vt,
                const float* __restrict__ vox, const float* __restrict__ z,
                _Float16* __restrict__ cat, int M)
{
    int i = blockIdx.x * 256 + threadIdx.x;
    if (i >= M * 288) return;
    int m = i / 288, c = i - m * 288;
    _Float16 v;
    if (c < 128)       v = enc[vt[m] * 128 + c];
    else if (c < 138)  v = (_Float16)vox[m * 10 + (c - 128)];
    else if (c < 266)  v = (_Float16)z[(size_t)m * 128 + (c - 138)];
    else               v = (_Float16)0.f;
    cat[i] = v;
}

__global__ __launch_bounds__(256)
void build_dec_tail(const _Float16* __restrict__ enc, const int* __restrict__ vt,
                    const float* __restrict__ vox, const float* __restrict__ z,
                    _Float16* __restrict__ cat, int M)
{
    int i = blockIdx.x * 256 + threadIdx.x;
    if (i >= M * 288) return;
    int m = i / 288, c = i - m * 288;   // maps to dec_cat column 1024+c
    _Float16 v;
    if (c < 128)       v = enc[vt[m] * 128 + c];
    else if (c < 138)  v = (_Float16)vox[m * 10 + (c - 128)];
    else if (c < 266)  v = (_Float16)z[(size_t)m * 128 + (c - 138)];
    else               v = (_Float16)0.f;
    cat[(size_t)m * 1312 + 1024 + c] = v;
}

// ---------------------------------------------------------------------------
// Final linear 64->8 + gumbel softmax + straight-through (all three outputs)
// ---------------------------------------------------------------------------
__global__ __launch_bounds__(256)
void final_head(const _Float16* __restrict__ D, int ldd,
                const float* __restrict__ Wl, const float* __restrict__ bl,
                const float* __restrict__ gn, float* __restrict__ out, int M)
{
    int m = blockIdx.x * 256 + threadIdx.x;
    if (m >= M) return;
    float l[8];
    #pragma unroll
    for (int j = 0; j < 8; ++j) l[j] = bl[j];
    for (int k = 0; k < 64; ++k) {
        float d = (float)D[(size_t)m * ldd + k];
        #pragma unroll
        for (int j = 0; j < 8; ++j) l[j] += d * Wl[k * 8 + j];
    }
    float pert[8], mx = -3.4e38f;
    #pragma unroll
    for (int j = 0; j < 8; ++j) { pert[j] = l[j] + gn[m * 8 + j]; mx = fmaxf(mx, pert[j]); }
    float ex[8], s = 0.f;
    #pragma unroll
    for (int j = 0; j < 8; ++j) { ex[j] = expf(pert[j] - mx); s += ex[j]; }
    float inv = 1.f / s;
    float soft[8]; int am = 0; float best = -1.f;
    #pragma unroll
    for (int j = 0; j < 8; ++j) {
        soft[j] = ex[j] * inv;
        if (soft[j] > best) { best = soft[j]; am = j; }
    }
    size_t base = (size_t)m * 8, MB = (size_t)M * 8;
    #pragma unroll
    for (int j = 0; j < 8; ++j) {
        float hard = (j == am) ? 1.f : 0.f;
        out[base + j]          = l[j];                       // logits
        out[MB + base + j]     = (hard - soft[j]) + soft[j]; // straight-through
        out[2 * MB + base + j] = soft[j];                    // soft
    }
}

// ===========================================================================
extern "C" void kernel_launch(void* const* d_in, const int* in_sizes, int n_in,
                              void* d_out, int out_size, void* d_ws, size_t ws_size,
                              hipStream_t stream)
{
    auto F = [&](int i) { return (const float*)d_in[i]; };
    auto I = [&](int i) { return (const int*)d_in[i]; };

    const int NL = in_sizes[0] / 13;          // 120
    const int N  = in_sizes[1] / 10;          // 50000
    const int NE = in_sizes[78] / 2;          // 400000

    const float* local_x  = F(0);
    const float* voxel_x  = F(1);
    const float* z        = F(2);
    const float* gnoise   = F(3);
    const int*   local_t  = I(76);
    const int*   voxel_t  = I(77);
    const int*   e_src    = I(78);
    const int*   e_dst    = e_src + NE;

    // ---- workspace carve-out ----
    char* ws = (char*)d_ws;
    size_t off = 0;
    auto alloc = [&](size_t bytes) -> void* {
        void* p = ws + off;
        off = (off + bytes + 255) & ~(size_t)255;
        return p;
    };

    // transposed f16 weights
    const int mlpK[3]  = {266, 512, 512}, mlpKp[3] = {288, 512, 512}, mlpN[3] = {512, 512, 512};
    const int cvK[6]   = {512, 256, 128, 64, 128, 256};
    const int cvN[6]   = {256, 128, 64, 128, 256, 512};
    const int decK[4]  = {1290, 512, 256, 128}, decKp[4] = {1312, 512, 256, 128};
    const int decN[4]  = {512, 256, 128, 64};

    _Float16* wt_mlp[3]; _Float16* wt_cv[6]; _Float16* wt_dec[4];
    for (int l = 0; l < 3; ++l) wt_mlp[l] = (_Float16*)alloc((size_t)mlpN[l] * mlpKp[l] * 2);
    for (int l = 0; l < 6; ++l) wt_cv[l]  = (_Float16*)alloc((size_t)cvN[l] * cvK[l] * 2);
    for (int l = 0; l < 4; ++l) wt_dec[l] = (_Float16*)alloc((size_t)decN[l] * decKp[l] * 2);

    float*     tmean   = (float*)alloc(NUM_TYPES * 13 * 4);
    int*       counts  = (int*)alloc(NUM_TYPES * 4);
    int*       deg     = (int*)alloc((size_t)N * 4);
    float*     dis     = (float*)alloc((size_t)N * 4);
    _Float16*  enc     = (_Float16*)alloc(NUM_TYPES * 128 * 2);
    _Float16*  cat0    = (_Float16*)alloc((size_t)N * 288 * 2);
    _Float16*  act     = (_Float16*)alloc((size_t)N * 512 * 2);
    float*     y       = (float*)alloc((size_t)N * 512 * 4);
    float*     gout    = (float*)alloc((size_t)N * 512 * 4);
    _Float16*  dcat    = (_Float16*)alloc((size_t)N * 1312 * 2);
    float*     csum    = (float*)alloc(512 * 4);
    float*     csq     = (float*)alloc(512 * 4);

    auto blk = [](long long n) { return (unsigned)((n + 255) / 256); };

    // ---- weight conversion ----
    for (int l = 0; l < 3; ++l)
        wt_convert<<<blk((long long)mlpN[l] * mlpKp[l]), 256, 0, stream>>>(
            F(16 + 4 * l), wt_mlp[l], mlpK[l], mlpN[l], mlpKp[l]);
    for (int l = 0; l < 6; ++l)
        wt_convert<<<blk((long long)cvN[l] * cvK[l]), 256, 0, stream>>>(
            F(28 + 5 * l), wt_cv[l], cvK[l], cvN[l], cvK[l]);
    for (int l = 0; l < 4; ++l)
        wt_convert<<<blk((long long)decN[l] * decKp[l]), 256, 0, stream>>>(
            F(58 + 4 * l), wt_dec[l], decK[l], decN[l], decKp[l]);

    // ---- type means, voxel type counts, degrees ----
    type_mean<<<1, 128, 0, stream>>>(local_x, local_t, tmean, NL);
    cnt_init<<<1, 32, 0, stream>>>(counts);
    vox_count<<<blk(N), 256, 0, stream>>>(voxel_t, counts, N);
    deg_init<<<blk(N), 256, 0, stream>>>(deg, N);
    deg_accum<<<blk(NE), 256, 0, stream>>>(e_dst, deg, NE);
    deg_finish<<<blk(N), 256, 0, stream>>>(deg, dis, N);

    // ---- local encoder on 8 rows ----
    lge_forward<<<1, 128, 0, stream>>>(tmean, counts, N,
        F(4), F(5), F(6), F(7), F(8), F(9), F(10), F(11), F(12), F(13), F(14), F(15), enc);

    // ---- MLP input concat ----
    build_cat0<<<blk((long long)N * 288), 256, 0, stream>>>(enc, voxel_t, voxel_x, z, cat0, N);

    const unsigned mB = (unsigned)((N + 63) / 64);

    // ---- MLP: 3 x (GEMM -> stats -> BN+LeakyReLU) ----
    for (int l = 0; l < 3; ++l) {
        const _Float16* Ain = (l == 0) ? cat0 : act;
        int lda = (l == 0) ? 288 : 512;
        gemm_wmma_f16<<<dim3(512 / 64, mB), 128, 0, stream>>>(
            Ain, lda, wt_mlp[l], mlpKp[l], F(16 + 4 * l + 1), y, 512, N, mlpKp[l]);
        col_reduce<<<512 / 64, 256, 0, stream>>>(y, 512, N, csum, csq);
        bool last = (l == 2);
        norm_act<<<blk((long long)N * 512), 256, 0, stream>>>(
            y, 512, N, csum, csq, F(16 + 4 * l + 2), F(16 + 4 * l + 3), nullptr, 0,
            act, 512, 0, last ? dcat : nullptr, 1312, 512);  // last MLP also -> dec concat "x"
    }

    // ---- GCN: 6 x (GEMM -> self+bias -> edge scatter -> stats -> GraphNorm+ReLU)
    for (int l = 0; l < 6; ++l) {
        int Cin = cvK[l], Cout = cvN[l];
        gemm_wmma_f16<<<dim3(Cout / 64, mB), 128, 0, stream>>>(
            act, Cin, wt_cv[l], Cin, nullptr, y, Cout, N, Cin);
        gcn_self<<<blk((long long)N * Cout), 256, 0, stream>>>(
            y, dis, F(28 + 5 * l + 1), gout, Cout, N);
        gcn_scatter<<<dim3((unsigned)NE, (Cout + 255) / 256), 256, 0, stream>>>(
            y, e_src, e_dst, dis, gout, Cout);
        col_reduce<<<(Cout + 63) / 64, 256, 0, stream>>>(gout, Cout, N, csum, csq);
        bool last = (l == 5);
        norm_act<<<blk((long long)N * Cout), 256, 0, stream>>>(
            gout, Cout, N, csum, csq, F(28 + 5 * l + 2), F(28 + 5 * l + 3), F(28 + 5 * l + 4), 1,
            act, Cout, 0, last ? dcat : nullptr, 1312, 0);   // last GCN -> dec concat "encoded"
    }

    // ---- decoder concat tail + 4 x (GEMM -> BN+LeakyReLU) ----
    build_dec_tail<<<blk((long long)N * 288), 256, 0, stream>>>(enc, voxel_t, voxel_x, z, dcat, N);
    for (int l = 0; l < 4; ++l) {
        const _Float16* Ain = (l == 0) ? dcat : act;
        int lda = (l == 0) ? 1312 : decN[l - 1];
        int Cout = decN[l];
        gemm_wmma_f16<<<dim3(Cout / 64, mB), 128, 0, stream>>>(
            Ain, lda, wt_dec[l], decKp[l], F(58 + 4 * l + 1), y, Cout, N, decKp[l]);
        col_reduce<<<(Cout + 63) / 64, 256, 0, stream>>>(y, Cout, N, csum, csq);
        norm_act<<<blk((long long)N * Cout), 256, 0, stream>>>(
            y, Cout, N, csum, csq, F(58 + 4 * l + 2), F(58 + 4 * l + 3), nullptr, 0,
            act, Cout, 0, nullptr, 0, 0);
    }

    // ---- head: 64->8 linear + gumbel softmax + straight-through ----
    final_head<<<blk(N), 256, 0, stream>>>(act, 64, F(74), F(75), gnoise, (float*)d_out, N);
    (void)n_in; (void)out_size; (void)ws_size;
}
